// MultiHeadAttention_64441689309442
// MI455X (gfx1250) — matrix operand order repaired and checked by reading, not compiled, not run on hardware
//
#include <hip/hip_runtime.h>
#include <math.h>
#include <stdint.h>

// ---------------------------------------------------------------------------
// MI455X (gfx1250) multi-head attention, bf16 WMMA + f32 accumulate.
// Compute-bound (~69 GFLOP vs ~50MB I/O -> ~1400 FLOP/byte): everything
// targets v_wmma_f32_16x16x32_bf16. Flash-attention streaming softmax avoids
// the 1GB logits tensor; K/V tiles are staged into LDS once per block
// (8 waves share them) via double-buffered async copies when available.
// ---------------------------------------------------------------------------

typedef unsigned short ushort_t;
typedef __attribute__((ext_vector_type(16))) __bf16    bf16x16;
typedef __attribute__((ext_vector_type(8)))  float     floatx8;
typedef __attribute__((ext_vector_type(8)))  ushort_t  ushortx8;
typedef __attribute__((ext_vector_type(16))) ushort_t  ushortx16;
typedef int v4i_t __attribute__((vector_size(16)));    // matches builtin param

#define AS1 __attribute__((address_space(1)))
#define AS3 __attribute__((address_space(3)))

#if defined(__has_builtin)
#if __has_builtin(__builtin_amdgcn_global_load_async_to_lds_b128) && \
    __has_builtin(__builtin_amdgcn_s_wait_asynccnt)
#define HAVE_ASYNC_LDS 1
#endif
#endif
#ifndef HAVE_ASYNC_LDS
#define HAVE_ASYNC_LDS 0
#endif

union Frag16 {
    ushort_t  s[16];
    ushortx8  h[2];
    ushortx16 w;
    bf16x16   v;
};

__device__ __forceinline__ ushort_t f2bf(float f) {
    unsigned u = __float_as_uint(f);
    u += 0x7fffu + ((u >> 16) & 1u);   // round-to-nearest-even
    return (ushort_t)(u >> 16);
}

// A fragment: 16x32 (MxK), bf16, from row-major source (K contiguous).
// ISA layout (16-bit A 16x32): lane m = lane&15, half = lane>>4;
//   elems 0..7  -> K = half*8 + e ; elems 8..15 -> K = 16 + half*8 + (e-8)
// => two contiguous 16B loads per lane.
__device__ __forceinline__ bf16x16 load_a_bf16(const ushort_t* base, int ld, int lane) {
    const int m = lane & 15, hf = lane >> 4;
    const ushort_t* p = base + m * ld + hf * 8;
    Frag16 f;
    f.h[0] = *(const ushortx8*)(p);
    f.h[1] = *(const ushortx8*)(p + 16);
    return f.v;
}

// B fragment: 32x16 (KxN), bf16, loaded from B^T stored row-major (N rows of
// K-contiguous data). Lane n = lane&15, half = lane>>4; elems 0..7 -> K = half*8 + e;
// elems 8..15 -> K = 16 + half*8 + (e-8)  => two contiguous 16B loads per lane.
__device__ __forceinline__ bf16x16 load_bt_bf16(const ushort_t* base, int ld, int lane) {
    const int n = lane & 15, hf = lane >> 4;
    const ushort_t* p = base + n * ld + hf * 8;
    Frag16 f;
    f.h[0] = *(const ushortx8*)(p);
    f.h[1] = *(const ushortx8*)(p + 16);
    return f.v;
}

__device__ __forceinline__ floatx8 wmma_bf16(bf16x16 a, bf16x16 b, floatx8 c) {
    return __builtin_amdgcn_wmma_f32_16x16x32_bf16(false, a, false, b, (short)0, c, false, false);
}

// ---------------------------------------------------------------------------
__global__ void cvt_f32_to_bf16_kernel(const float* __restrict__ in,
                                       ushort_t* __restrict__ out, int n) {
    int i = blockIdx.x * blockDim.x + threadIdx.x;
    if (i < n) out[i] = f2bf(in[i]);
}

// C[MxN] = A[MxK] * Bt[NxK]^T. One wave -> 32x64 output tile (2 A-frags
// share 4 B-frags: 12 b128 loads per 8 WMMA). 8 waves per block.
__global__ void __launch_bounds__(256)
gemm_bf16_wmma_kernel(const ushort_t* __restrict__ A, const ushort_t* __restrict__ Bt,
                      float* __restrict__ C, int M, int N, int K) {
    const int lane = threadIdx.x & 31;
    const int gw   = blockIdx.x * 8 + (threadIdx.x >> 5);
    const int ntiles = N >> 6;
    const int m0 = (gw / ntiles) * 32;
    const int n0 = (gw % ntiles) * 64;
    if (m0 >= M) return;

    const floatx8 z = {0.f,0.f,0.f,0.f,0.f,0.f,0.f,0.f};
    floatx8 acc[2][4] = {{z,z,z,z},{z,z,z,z}};

    for (int k0 = 0; k0 < K; k0 += 32) {
        bf16x16 a0 = load_a_bf16(A + (size_t)m0 * K + k0, K, lane);
        bf16x16 a1 = load_a_bf16(A + (size_t)(m0 + 16) * K + k0, K, lane);
        #pragma unroll
        for (int j = 0; j < 4; ++j) {
            bf16x16 b = load_bt_bf16(Bt + (size_t)(n0 + 16 * j) * K + k0, K, lane);
            acc[0][j] = wmma_bf16(a0, b, acc[0][j]);
            acc[1][j] = wmma_bf16(a1, b, acc[1][j]);
        }
    }

    const int mr = (lane >> 4) * 8;  // C/D layout: VGPR r -> row r + 8*(lane>>4)
    const int n  = lane & 15;
    #pragma unroll
    for (int t = 0; t < 2; ++t)
        #pragma unroll
        for (int r = 0; r < 8; ++r) {
            float* crow = C + (size_t)(m0 + 16 * t + mr + r) * N + n0 + n;
            crow[ 0] = acc[t][0][r];
            crow[16] = acc[t][1][r];
            crow[32] = acc[t][2][r];
            crow[48] = acc[t][3][r];
        }
}

// RoPE on q,k; split qkv[B,L,3,H,hd] (f32) into bf16 q,k:[B,H,L,hd] and
// v transposed: [B,H,hd,L] (so PV's B-operand is K-contiguous).
__global__ void rope_split_kernel(const float* __restrict__ qkv,
                                  ushort_t* __restrict__ qb,
                                  ushort_t* __restrict__ kb,
                                  ushort_t* __restrict__ vt,
                                  int B, int L, int H, int hd) {
    int t = blockIdx.x * blockDim.x + threadIdx.x;
    const int nh = hd >> 1;
    if (t >= B * H * L * nh) return;
    const int i = t % nh;  t /= nh;
    const int l = t % L;   t /= L;
    const int h = t % H;   t /= H;
    const int b = t;
    const int D = H * hd;

    float inv = __expf(-((float)(2 * i) / (float)hd) * 9.210340371976184f); // 10000^(-2i/hd)
    float ang = (float)l * inv;
    float s, c;
    sincosf(ang, &s, &c);

    const float* base = qkv + (size_t)(b * L + l) * (3 * D);
    const size_t qo = ((size_t)((b * H + h) * L) + l) * hd;

    float x1 = base[h * hd + 2 * i], x2 = base[h * hd + 2 * i + 1];
    qb[qo + 2 * i]     = f2bf(x1 * c - x2 * s);
    qb[qo + 2 * i + 1] = f2bf(x1 * s + x2 * c);

    x1 = base[D + h * hd + 2 * i];  x2 = base[D + h * hd + 2 * i + 1];
    kb[qo + 2 * i]     = f2bf(x1 * c - x2 * s);
    kb[qo + 2 * i + 1] = f2bf(x1 * s + x2 * c);

    const float v0 = base[2 * D + h * hd + 2 * i];
    const float v1 = base[2 * D + h * hd + 2 * i + 1];
    const size_t vo = ((size_t)((b * H + h) * hd)) * L + l;
    vt[vo + (size_t)(2 * i) * L]     = f2bf(v0);
    vt[vo + (size_t)(2 * i + 1) * L] = f2bf(v1);
}

// Flash attention. One wave per 16-row q tile; all 8 waves of a block share
// one (b,h) (128 q-tiles per head, 8 per block), so K/V kv-blocks of 32 are
// staged once per block into double-buffered LDS (async DMA when available)
// instead of 8x redundant global loads.
__global__ void __launch_bounds__(256)
flash_attn_wmma_kernel(const ushort_t* __restrict__ qb, const ushort_t* __restrict__ kb,
                       const ushort_t* __restrict__ vt, ushort_t* __restrict__ ao,
                       int B, int L, int H, int hd) {
    __shared__ ushort_t kT[2][32 * 64];   // K tile: 32 kv rows x 64 d   (4KB x2)
    __shared__ ushort_t vT[2][64 * 32];   // V^T tile: 64 d rows x 32 kv (4KB x2)
    __shared__ ushort_t Pt[8][16 * 32];   // per-wave P tile             (8KB)

    const int tid  = threadIdx.x;
    const int lane = tid & 31;
    const int wv   = tid >> 5;
    const int gw   = blockIdx.x * 8 + wv;
    const int qtiles = L >> 4;
    const int q0 = (gw % qtiles) * 16;
    const int bh = gw / qtiles;          // b*H + h (uniform across the block)
    const int h  = bh % H;
    const int b  = bh / H;
    const float scale = 0.125f;          // hd^-0.5, hd=64

    const ushort_t* qbase = qb + (size_t)bh * L * hd;
    const ushort_t* kbase = kb + (size_t)bh * L * hd;
    const ushort_t* vbase = vt + (size_t)bh * hd * L;

    // block-cooperative staging: 256 threads x 16B each per tile
    const int krow = tid >> 3, kcol = (tid & 7) * 8;   // K: 32 rows x 128B
    const int vrow = tid >> 2, vcol = (tid & 3) * 8;   // V: 64 rows x 64B

    auto stage = [&](int buf, int kv0) {
        const ushort_t* gk = kbase + (size_t)(kv0 + krow) * hd + kcol;
        const ushort_t* gv = vbase + (size_t)vrow * L + kv0 + vcol;
        ushort_t* lk = &kT[buf][krow * 64 + kcol];
        ushort_t* lv = &vT[buf][vrow * 32 + vcol];
#if HAVE_ASYNC_LDS
        __builtin_amdgcn_global_load_async_to_lds_b128(
            (AS1 v4i_t*)((void*)gk), (AS3 v4i_t*)lk, 0, 0);
        __builtin_amdgcn_global_load_async_to_lds_b128(
            (AS1 v4i_t*)((void*)gv), (AS3 v4i_t*)lv, 0, 0);
#else
        *(ushortx8*)lk = *(const ushortx8*)gk;
        *(ushortx8*)lv = *(const ushortx8*)gv;
#endif
    };

    const bf16x16 aQ0 = load_a_bf16(qbase + (size_t)q0 * hd, hd, lane);      // d 0..31
    const bf16x16 aQ1 = load_a_bf16(qbase + (size_t)q0 * hd + 32, hd, lane); // d 32..63

    const floatx8 z = {0.f,0.f,0.f,0.f,0.f,0.f,0.f,0.f};
    floatx8 o0 = z, o1 = z, o2 = z, o3 = z;
    float mrow[8], lrow[8];
    #pragma unroll
    for (int r = 0; r < 8; ++r) { mrow[r] = -1e30f; lrow[r] = 0.f; }

    const int mr = (lane >> 4) * 8;
    const int n  = lane & 15;
    ushort_t* Pw = &Pt[wv][0];

    stage(0, 0);   // prologue

    for (int kv0 = 0; kv0 < L; kv0 += 32) {
        const int cur = (kv0 >> 5) & 1;
#if HAVE_ASYNC_LDS
        __builtin_amdgcn_s_wait_asynccnt(0);   // my copies for buffer `cur` done
#endif
        __syncthreads();                       // everyone's copies visible
        if (kv0 + 32 < L) stage(cur ^ 1, kv0 + 32);   // overlap DMA with compute

        // ---- S = Q K^T (16 x 32) from LDS K tile ----
        floatx8 s0 = z, s1 = z;
        bf16x16 bk;
        bk = load_bt_bf16(&kT[cur][ 0 * 64 +  0], 64, lane); s0 = wmma_bf16(aQ0, bk, s0);
        bk = load_bt_bf16(&kT[cur][ 0 * 64 + 32], 64, lane); s0 = wmma_bf16(aQ1, bk, s0);
        bk = load_bt_bf16(&kT[cur][16 * 64 +  0], 64, lane); s1 = wmma_bf16(aQ0, bk, s1);
        bk = load_bt_bf16(&kT[cur][16 * 64 + 32], 64, lane); s1 = wmma_bf16(aQ1, bk, s1);

        // ---- online softmax (row stats live across a 16-lane half) ----
        float alpha[8];
        #pragma unroll
        for (int r = 0; r < 8; ++r) {
            float x0 = s0[r] * scale;
            float x1 = s1[r] * scale;
            float mx = fmaxf(x0, x1);
            mx = fmaxf(mx, __shfl_xor(mx, 1, 32));
            mx = fmaxf(mx, __shfl_xor(mx, 2, 32));
            mx = fmaxf(mx, __shfl_xor(mx, 4, 32));
            mx = fmaxf(mx, __shfl_xor(mx, 8, 32));
            float mnew = fmaxf(mrow[r], mx);
            float a = __expf(mrow[r] - mnew);
            mrow[r] = mnew;
            float p0 = __expf(x0 - mnew);
            float p1 = __expf(x1 - mnew);
            float rs = p0 + p1;
            rs += __shfl_xor(rs, 1, 32);
            rs += __shfl_xor(rs, 2, 32);
            rs += __shfl_xor(rs, 4, 32);
            rs += __shfl_xor(rs, 8, 32);
            lrow[r] = a * lrow[r] + rs;
            alpha[r] = a;
            s0[r] = p0;
            s1[r] = p1;
        }
        #pragma unroll
        for (int r = 0; r < 8; ++r) {
            o0[r] *= alpha[r]; o1[r] *= alpha[r];
            o2[r] *= alpha[r]; o3[r] *= alpha[r];
        }

        // ---- P (C layout) -> per-wave LDS slab -> A fragment ----
        // Wave-private slab: per-wave DS ops are in-order, no barrier needed.
        #pragma unroll
        for (int r = 0; r < 8; ++r) {
            Pw[(mr + r) * 32 + n]      = f2bf(s0[r]);
            Pw[(mr + r) * 32 + 16 + n] = f2bf(s1[r]);
        }
        bf16x16 aP = load_a_bf16(Pw, 32, lane);

        // ---- O += P V from LDS V^T tile ----
        bf16x16 bv;
        bv = load_bt_bf16(&vT[cur][ 0 * 32], 32, lane); o0 = wmma_bf16(aP, bv, o0);
        bv = load_bt_bf16(&vT[cur][16 * 32], 32, lane); o1 = wmma_bf16(aP, bv, o1);
        bv = load_bt_bf16(&vT[cur][32 * 32], 32, lane); o2 = wmma_bf16(aP, bv, o2);
        bv = load_bt_bf16(&vT[cur][48 * 32], 32, lane); o3 = wmma_bf16(aP, bv, o3);
    }

    // normalize and write attn output as bf16 in [B, L, H*hd] (GEMM2 input)
    const int D = H * hd;
    #pragma unroll
    for (int r = 0; r < 8; ++r) {
        const float inv = 1.0f / lrow[r];
        ushort_t* row = ao + (size_t)(b * L + q0 + mr + r) * D + h * hd + n;
        row[ 0] = f2bf(o0[r] * inv);
        row[16] = f2bf(o1[r] * inv);
        row[32] = f2bf(o2[r] * inv);
        row[48] = f2bf(o3[r] * inv);
    }
}

// ---------------------------------------------------------------------------
extern "C" void kernel_launch(void* const* d_in, const int* in_sizes, int n_in,
                              void* d_out, int out_size, void* d_ws, size_t ws_size,
                              hipStream_t stream) {
    const float* x    = (const float*)d_in[0];   // [B,L,D]
    const float* Wqkv = (const float*)d_in[1];   // [3D,D]
    const float* Wout = (const float*)d_in[2];   // [D,D]
    float* out = (float*)d_out;                  // [B,L,D] f32

    constexpr int B = 2, L = 2048, D = 1024, H = 16, HD = 64;
    constexpr int M  = B * L;       // 4096
    constexpr int N1 = 3 * D;       // 3072

    ushort_t* xb  = (ushort_t*)d_ws;                 // M*D       bf16
    ushort_t* wqb = xb  + (size_t)M * D;             // N1*D      bf16
    ushort_t* wob = wqb + (size_t)N1 * D;            // D*D       bf16
    ushort_t* qb  = wob + (size_t)D * D;             // B*H*L*HD  bf16
    ushort_t* kb  = qb  + (size_t)B * H * L * HD;
    ushort_t* vtb = kb  + (size_t)B * H * L * HD;    // [B,H,HD,L]
    ushort_t* aob = vtb + (size_t)B * H * L * HD;    // M*D bf16 (attn out)
    float*    qkv = (float*)(aob + (size_t)M * D);   // M*N1 f32

    // 1) f32 -> bf16 operand conversion
    cvt_f32_to_bf16_kernel<<<(M * D  + 255) / 256, 256, 0, stream>>>(x,    xb,  M * D);
    cvt_f32_to_bf16_kernel<<<(N1 * D + 255) / 256, 256, 0, stream>>>(Wqkv, wqb, N1 * D);
    cvt_f32_to_bf16_kernel<<<(D * D  + 255) / 256, 256, 0, stream>>>(Wout, wob, D * D);

    // 2) qkv = x @ W_qkv^T   (M x N1, K = D); 32x64 tiles per wave
    gemm_bf16_wmma_kernel<<<(M / 32) * (N1 / 64) / 8, 256, 0, stream>>>(xb, wqb, qkv, M, N1, D);

    // 3) RoPE + head split (+ V transpose)
    const int rt = B * H * L * (HD / 2);
    rope_split_kernel<<<(rt + 255) / 256, 256, 0, stream>>>(qkv, qb, kb, vtb, B, L, H, HD);

    // 4) flash attention: one wave per 16-row q tile, 8 waves/block share (b,h)
    const int qwaves = B * H * (L / 16);    // 4096
    flash_attn_wmma_kernel<<<qwaves / 8, 256, 0, stream>>>(qb, kb, vtb, aob, B, L, H, HD);

    // 5) out = attn @ W_out^T  (M x D, K = D), f32 result
    gemm_bf16_wmma_kernel<<<(M / 32) * (D / 64) / 8, 256, 0, stream>>>(aob, wob, out, M, D, D);
}
